// MistralPagedWrapper_89618787598499
// MI455X (gfx1250) — matrix operand order repaired and checked by reading, not compiled, not run on hardware
//
#include <hip/hip_runtime.h>
#include <hip/hip_bf16.h>
#include <math.h>

// ---------------- problem constants (from reference) ----------------
constexpr int Lc   = 4;
constexpr int Bc   = 64;
constexpr int Qc   = 4;
constexpr int Sc   = 512;
constexpr int PSc  = 16;
constexpr int Pc   = 32;
constexpr int DMc  = 2048;
constexpr int Hc   = 32;
constexpr int HKVc = 8;
constexpr int HDc  = 64;
constexpr int DFFc = 4096;
constexpr int Vc   = 32000;
constexpr int BQ   = Bc * Qc;              // 256 token rows
constexpr float EPSc = 1e-5f;

typedef __attribute__((ext_vector_type(16))) __bf16 v16bf;
typedef __attribute__((ext_vector_type(8)))  float  v8f;

// ---------------- embedding gather ----------------
__global__ void k_embed(const int* __restrict__ ids, const float* __restrict__ embed,
                        float* __restrict__ hidden) {
  const int row = blockIdx.x;                  // 0..BQ-1
  const int tok = ids[row];
  const float* src = embed + (size_t)tok * DMc;
  float* dst = hidden + (size_t)row * DMc;
  for (int d = threadIdx.x; d < DMc; d += blockDim.x) dst[d] = src[d];
}

// ---------------- RMSNorm (one block per row) ----------------
__global__ void k_rmsnorm(const float* __restrict__ x, const float* __restrict__ w,
                          float* __restrict__ y) {
  const int row = blockIdx.x;
  const float* xr = x + (size_t)row * DMc;
  float s = 0.f;
  for (int i = threadIdx.x; i < DMc; i += 256) { float v = xr[i]; s += v * v; }
  for (int off = 16; off; off >>= 1) s += __shfl_down(s, off, 32);
  __shared__ float red[8];
  __shared__ float sinv;
  if ((threadIdx.x & 31) == 0) red[threadIdx.x >> 5] = s;
  __syncthreads();
  if (threadIdx.x == 0) {
    float t = 0.f;
    #pragma unroll
    for (int i = 0; i < 8; ++i) t += red[i];
    sinv = rsqrtf(t / (float)DMc + EPSc);
  }
  __syncthreads();
  const float inv = sinv;
  float* yr = y + (size_t)row * DMc;
  for (int i = threadIdx.x; i < DMc; i += 256) yr[i] = xr[i] * inv * w[i];
}

// ---------------- WMMA bf16 GEMM: C[M,N] = A[M,K] @ B[K,N] (+ residual) ----------------
// Templated on (K, N) so every strided B load uses a compile-time immediate offset.
// grid = (N/256, BQ/16), block = 256 threads = 8 waves.
// Wave w owns two 16x16 tiles at columns n0 = 256*bx + 16w and n0+128.
// A chunk (16 x KCH fp32) staged in LDS via async-to-LDS; inner K loop is barrier-free.
constexpr int KCH = 512;                       // 16*512*4B = 32 KB LDS

template <int K, int N, bool ADD>
__global__ void k_gemm(const float* __restrict__ A, const float* __restrict__ Bw,
                       const float* __restrict__ addsrc, float* __restrict__ C) {
  const int tid  = threadIdx.x;
  const int wave = tid >> 5;
  const int lane = tid & 31;
  const int g    = lane >> 4;        // K-half select
  const int hl   = lane & 15;
  const int m0   = blockIdx.y * 16;
  const int n0   = blockIdx.x * 256 + wave * 16;

  __shared__ float As[16 * KCH];     // fp32 A chunk, row-major [16][KCH]

  v8f acc0 = {}, acc1 = {};
  // running B pointer: column n0+hl, row 16*g; advances 32 rows per K-tile
  const float* bp = Bw + (size_t)(16 * g) * N + (n0 + hl);

  for (int kc = 0; kc < K; kc += KCH) {
    // ---- stage A chunk (16 x KCH fp32) via async global->LDS b128 copies ----
    {
      constexpr int F4R = KCH / 4;             // float4 per row
      #pragma unroll
      for (int p = 0; p < 16 * F4R / 256; ++p) {
        const int idx4 = tid + p * 256;
        const int r  = idx4 / F4R;
        const int c4 = idx4 % F4R;
        const unsigned lds = (unsigned)(size_t)(const void*)&As[r * KCH + c4 * 4];
        const float* gp = A + (size_t)(m0 + r) * K + kc + c4 * 4;
        asm volatile("global_load_async_to_lds_b128 %0, %1, off"
                     :: "v"(lds), "v"(gp) : "memory");
      }
      asm volatile("s_wait_asynccnt 0" ::: "memory");
      __syncthreads();
    }

    // ---- barrier-free inner loop: stream B, two WMMAs per 32-K tile ----
    #pragma unroll 2
    for (int kt = 0; kt < KCH; kt += 32) {
      // A fragment (shared by both output tiles): row hl, K {16g..16g+15}
      const float* ap = &As[hl * KCH + kt + 16 * g];
      v16bf af;
      #pragma unroll
      for (int j = 0; j < 8; ++j) {
        af[j]     = (__bf16)ap[j];
        af[8 + j] = (__bf16)ap[8 + j];
      }

      // B fragments: 16 rows at compile-time offsets j*N, columns n0+hl and +128
      __builtin_prefetch(bp + (size_t)64 * N, 0, 1);     // two tiles ahead, speculative
      v16bf bf0, bf1;
      #pragma unroll
      for (int j = 0; j < 16; ++j) {
        bf0[j] = (__bf16)bp[j * N];
        bf1[j] = (__bf16)bp[j * N + 128];
      }
      bp += (size_t)32 * N;

      acc0 = __builtin_amdgcn_wmma_f32_16x16x32_bf16(false, af, false, bf0,
                                                     (short)0, acc0, false, false);
      acc1 = __builtin_amdgcn_wmma_f32_16x16x32_bf16(false, af, false, bf1,
                                                     (short)0, acc1, false, false);
    }
    __syncthreads();                 // before re-staging As
  }

  // C layout: acc[r] -> row m0 + r + 8g, col n0 + hl (and +128)
  #pragma unroll
  for (int r = 0; r < 8; ++r) {
    const size_t o0 = (size_t)(m0 + r + 8 * g) * N + (n0 + hl);
    const size_t o1 = o0 + 128;
    float v0 = acc0[r], v1 = acc1[r];
    if constexpr (ADD) { v0 += addsrc[o0]; v1 += addsrc[o1]; }
    C[o0] = v0;
    C[o1] = v1;
  }
}

// ---------------- RoPE (one block of 32 lanes per (b,q,head)) ----------------
__global__ void k_rope(float* __restrict__ x, int nheads) {
  const int head_blk = blockIdx.x;                 // b*Q*nheads + qi*nheads + h
  const int qi = (head_blk / nheads) % Qc;
  float* p = x + (size_t)head_blk * HDc;
  const int j = threadIdx.x;                       // 0..31 (half = HD/2)
  const float inv = powf(10000.f, -(float)j / 32.f);
  const float ang = (float)(Sc + qi) * inv;
  const float c = __cosf(ang), s = __sinf(ang);
  const float x1 = p[j], x2 = p[j + 32];
  p[j]      = x1 * c - x2 * s;
  p[j + 32] = x2 * c + x1 * s;
}

// ---------------- paged attention: one block (128 thr) per (b,h,qi) ----------------
__global__ void k_attn(const float* __restrict__ qbuf, const float* __restrict__ kbuf,
                       const float* __restrict__ vbuf, const float* __restrict__ kcache,
                       const float* __restrict__ vcache, const int* __restrict__ req_id,
                       const int* __restrict__ page_table, float* __restrict__ outb) {
  const int idx = blockIdx.x;                // b*H*Q + h*Q + qi
  const int qi  = idx % Qc;
  const int h   = (idx / Qc) % Hc;
  const int b   = idx / (Qc * Hc);
  const int kvh = h / (Hc / HKVc);
  const int nk  = Sc + qi + 1;               // causal: keys 0..S+qi
  const int tid = threadIdx.x;

  __shared__ float qv[HDc];
  __shared__ float lg[Sc + Qc];
  __shared__ float red[4];
  __shared__ int   pgs[Pc];

  if (tid < HDc) qv[tid] = qbuf[((size_t)(b * Qc + qi) * Hc + h) * HDc + tid];
  if (tid >= HDc && tid < HDc + Pc) {
    const int rb = req_id[b];
    pgs[tid - HDc] = page_table[rb * Pc + (tid - HDc)];
  }
  __syncthreads();

  float lmax = -1e30f;
  for (int s = tid; s < nk; s += 128) {
    const float* kp;
    if (s < Sc) {
      kp = kcache + (((size_t)pgs[s >> 4] * PSc + (s & 15)) * HKVc + kvh) * HDc;
    } else {
      kp = kbuf + ((size_t)(b * Qc + (s - Sc)) * HKVc + kvh) * HDc;
    }
    float acc = 0.f;
    #pragma unroll 8
    for (int d = 0; d < HDc; ++d) acc += qv[d] * kp[d];
    acc *= 0.125f;                           // 1/sqrt(64)
    lg[s] = acc;
    lmax = fmaxf(lmax, acc);
  }
  for (int off = 16; off; off >>= 1) lmax = fmaxf(lmax, __shfl_down(lmax, off, 32));
  if ((tid & 31) == 0) red[tid >> 5] = lmax;
  __syncthreads();
  const float m = fmaxf(fmaxf(red[0], red[1]), fmaxf(red[2], red[3]));
  __syncthreads();                           // everyone has m before red is reused

  float lsum = 0.f;
  for (int s = tid; s < nk; s += 128) {
    float e = __expf(lg[s] - m);
    lg[s] = e;
    lsum += e;
  }
  for (int off = 16; off; off >>= 1) lsum += __shfl_down(lsum, off, 32);
  if ((tid & 31) == 0) red[tid >> 5] = lsum;
  __syncthreads();
  const float inv = 1.f / (red[0] + red[1] + red[2] + red[3]);

  if (tid < HDc) {
    float acc = 0.f;
    for (int s = 0; s < nk; ++s) {
      const float* vp;
      if (s < Sc) {
        vp = vcache + (((size_t)pgs[s >> 4] * PSc + (s & 15)) * HKVc + kvh) * HDc;
      } else {
        vp = vbuf + ((size_t)(b * Qc + (s - Sc)) * HKVc + kvh) * HDc;
      }
      acc += lg[s] * vp[tid];
    }
    outb[((size_t)(b * Qc + qi) * Hc + h) * HDc + tid] = acc * inv;
  }
}

// ---------------- silu(gate) * up, in place into gate ----------------
__global__ void k_silu_mul(float* __restrict__ gate, const float* __restrict__ up) {
  const size_t i = (size_t)blockIdx.x * blockDim.x + threadIdx.x;
  float g = gate[i];
  gate[i] = g * (1.f / (1.f + __expf(-g))) * up[i];
}

// ---------------- host orchestration ----------------
extern "C" void kernel_launch(void* const* d_in, const int* in_sizes, int n_in,
                              void* d_out, int out_size, void* d_ws, size_t ws_size,
                              hipStream_t stream) {
  (void)in_sizes; (void)n_in; (void)out_size; (void)ws_size;

  const int*   input_ids  = (const int*)  d_in[0];
  const int*   req_id     = (const int*)  d_in[1];
  const int*   page_table = (const int*)  d_in[2];
  const float* embed      = (const float*)d_in[3];
  const float* w_in_ln    = (const float*)d_in[4];
  const float* wq         = (const float*)d_in[5];
  const float* wk         = (const float*)d_in[6];
  const float* wv         = (const float*)d_in[7];
  const float* wo         = (const float*)d_in[8];
  const float* w_post_ln  = (const float*)d_in[9];
  const float* w_gate     = (const float*)d_in[10];
  const float* w_up       = (const float*)d_in[11];
  const float* w_down     = (const float*)d_in[12];
  const float* k_cache    = (const float*)d_in[13];
  const float* v_cache    = (const float*)d_in[14];
  const float* lm_head    = (const float*)d_in[15];
  float* out = (float*)d_out;

  float* ws     = (float*)d_ws;
  float* hidden = ws;                        // BQ x DM
  float* hnorm  = hidden + (size_t)BQ * DMc; // BQ x DM
  float* qbuf   = hnorm  + (size_t)BQ * DMc; // BQ x (H*HD)
  float* kbuf   = qbuf   + (size_t)BQ * Hc * HDc;    // BQ x (HKV*HD)
  float* vbuf   = kbuf   + (size_t)BQ * HKVc * HDc;  // BQ x (HKV*HD)
  float* attnb  = vbuf   + (size_t)BQ * HKVc * HDc;  // BQ x (H*HD)
  float* gateb  = attnb  + (size_t)BQ * Hc * HDc;    // BQ x DFF
  float* upb    = gateb  + (size_t)BQ * DFFc;        // BQ x DFF

  const size_t cacheL = (size_t)Bc * Pc * PSc * HKVc * HDc;  // elems per layer

  const dim3 blk(256);
  const dim3 gN512 (512  / 256, BQ / 16);
  const dim3 gN2048(2048 / 256, BQ / 16);
  const dim3 gN4096(4096 / 256, BQ / 16);
  const dim3 gNV   (Vc   / 256, BQ / 16);

  k_embed<<<BQ, 256, 0, stream>>>(input_ids, embed, hidden);

  for (int l = 0; l < Lc; ++l) {
    const float* wq_l = wq + (size_t)l * DMc * Hc * HDc;
    const float* wk_l = wk + (size_t)l * DMc * HKVc * HDc;
    const float* wv_l = wv + (size_t)l * DMc * HKVc * HDc;
    const float* wo_l = wo + (size_t)l * Hc * HDc * DMc;
    const float* wg_l = w_gate + (size_t)l * DMc * DFFc;
    const float* wu_l = w_up   + (size_t)l * DMc * DFFc;
    const float* wd_l = w_down + (size_t)l * DFFc * DMc;

    k_rmsnorm<<<BQ, 256, 0, stream>>>(hidden, w_in_ln + (size_t)l * DMc, hnorm);

    k_gemm<DMc, Hc * HDc,   false><<<gN2048, blk, 0, stream>>>(hnorm, wq_l, nullptr, qbuf);
    k_gemm<DMc, HKVc * HDc, false><<<gN512,  blk, 0, stream>>>(hnorm, wk_l, nullptr, kbuf);
    k_gemm<DMc, HKVc * HDc, false><<<gN512,  blk, 0, stream>>>(hnorm, wv_l, nullptr, vbuf);

    k_rope<<<Bc * Qc * Hc,   32, 0, stream>>>(qbuf, Hc);
    k_rope<<<Bc * Qc * HKVc, 32, 0, stream>>>(kbuf, HKVc);

    k_attn<<<Bc * Hc * Qc, 128, 0, stream>>>(qbuf, kbuf, vbuf,
                                             k_cache + (size_t)l * cacheL,
                                             v_cache + (size_t)l * cacheL,
                                             req_id, page_table, attnb);

    // hidden = hidden + attn @ wo
    k_gemm<Hc * HDc, DMc, true><<<gN2048, blk, 0, stream>>>(attnb, wo_l, hidden, hidden);

    k_rmsnorm<<<BQ, 256, 0, stream>>>(hidden, w_post_ln + (size_t)l * DMc, hnorm);

    k_gemm<DMc, DFFc, false><<<gN4096, blk, 0, stream>>>(hnorm, wg_l, nullptr, gateb);
    k_gemm<DMc, DFFc, false><<<gN4096, blk, 0, stream>>>(hnorm, wu_l, nullptr, upb);

    k_silu_mul<<<(BQ * DFFc) / 256, 256, 0, stream>>>(gateb, upb);

    // hidden = hidden + (silu*up) @ w_down
    k_gemm<DFFc, DMc, true><<<gN2048, blk, 0, stream>>>(gateb, wd_l, hidden, hidden);
  }

  // logits = hidden @ lm_head
  k_gemm<DMc, Vc, false><<<gNV, blk, 0, stream>>>(hidden, lm_head, nullptr, out);
}
